// ConvLayer_36395552866974
// MI455X (gfx1250) — matrix operand order repaired and checked by reading, not compiled, not run on hardware
//
#include <hip/hip_runtime.h>

typedef __attribute__((ext_vector_type(16))) __bf16    v16bf;
typedef __attribute__((ext_vector_type(8)))  float     v8f;
typedef __attribute__((ext_vector_type(8)))  unsigned  v8u;
typedef __attribute__((ext_vector_type(4)))  int       v4i;

constexpr int S_DIM  = 128;                  // site feature width
constexpr int BD_DIM = 64;                   // bond feature width
constexpr int K_DIM  = 320;                  // 2*S + BD
constexpr int TILE_M = 128;                  // edges per workgroup (16 per wave)
constexpr int PADK   = 336;                  // 320 + 16 pad -> 672B cols, 32B aligned
constexpr int LDSB   = 2 * S_DIM * PADK * 2; // 2 gates x 128 cols x PADK bf16 = 168 KB
constexpr int WDWORDS = LDSB / 4;            // packed-weight dwords in d_ws

// scalar fp32 -> bf16 RNE (cold path only: weight prep kernel)
static __device__ __forceinline__ __bf16 f2bf(float f) {
    unsigned u = __builtin_bit_cast(unsigned, f);
    unsigned r = (u + 0x7FFFu + ((u >> 16) & 1u)) >> 16;
    return __builtin_bit_cast(__bf16, (unsigned short)r);
}

// hot path: two fp32 -> packed bf16x2 (round-nearest ties-away) in 3 VALU
static __device__ __forceinline__ unsigned pack2bf(float lo, float hi) {
    unsigned ul = __builtin_bit_cast(unsigned, lo) + 0x8000u;
    unsigned uh = __builtin_bit_cast(unsigned, hi) + 0x8000u;
    // v_perm_b32: bytes {uh.b3, uh.b2, ul.b3, ul.b2} -> [lo_bf16 | hi_bf16<<16]
    return __builtin_amdgcn_perm(uh, ul, 0x07060302u);
}

__global__ void init_out_kernel(const float4* __restrict__ src,
                                float4* __restrict__ dst, int n4) {
    int i = blockIdx.x * blockDim.x + threadIdx.x;
    if (i < n4) dst[i] = src[i];
}

// one-time: both weight matrices -> bf16, transposed (col-major, k-contig),
// padded to PADK, packed as dwords -- the exact image each WG copies to LDS.
__global__ void prep_weights_kernel(const float* __restrict__ Wsig,
                                    const float* __restrict__ Wsoft,
                                    unsigned* __restrict__ wpack) {
    const int DW_COL = PADK / 2;             // 168 dwords per column
    int d = blockIdx.x * blockDim.x + threadIdx.x;
    if (d >= 2 * S_DIM * DW_COL) return;
    int g  = d / (S_DIM * DW_COL);
    int r  = d - g * (S_DIM * DW_COL);
    int n  = r / DW_COL;
    int kk = r - n * DW_COL;
    int k  = kk * 2;
    unsigned val = 0;
    if (k < K_DIM) {
        const float* W = g ? Wsoft : Wsig;
        unsigned lo = (unsigned)__builtin_bit_cast(unsigned short, f2bf(W[k * S_DIM + n]));
        unsigned hi = (unsigned)__builtin_bit_cast(unsigned short, f2bf(W[(k + 1) * S_DIM + n]));
        val = lo | (hi << 16);
    }
    wpack[d] = val;
}

__global__ void __launch_bounds__(256)
cgcnn_edge_wmma(const float* __restrict__ sites,
                const float* __restrict__ bonds,
                const unsigned* __restrict__ wpack,   // prepacked bf16 weights (d_ws)
                const float* __restrict__ bsig,
                const float* __restrict__ bsoft,
                const int*   __restrict__ idx1,
                const int*   __restrict__ idx2,
                float*       __restrict__ out,
                int E)
{
    extern __shared__ __bf16 smem[];   // Wt[gate][n][k], k-major, PADK stride
    const int tid = threadIdx.x;

    // ---- stage prepacked weights into LDS (raw 168 KB copy, async path) ----
#if __has_builtin(__builtin_amdgcn_global_load_async_to_lds_b128)
    {
        __attribute__((address_space(1))) v4i* gsrc =
            (__attribute__((address_space(1))) v4i*)wpack;
        __attribute__((address_space(3))) v4i* ldst =
            (__attribute__((address_space(3))) v4i*)smem;
        for (int i = tid; i < LDSB / 16; i += 256)
            __builtin_amdgcn_global_load_async_to_lds_b128(gsrc + i, ldst + i, 0, 0);
#if __has_builtin(__builtin_amdgcn_s_wait_asynccnt)
        __builtin_amdgcn_s_wait_asynccnt(0);
#else
        asm volatile("s_wait_asynccnt 0x0" ::: "memory");
#endif
    }
#else
    {
        const uint4* g4 = (const uint4*)wpack;
        uint4* s4 = (uint4*)smem;
        for (int i = tid; i < LDSB / 16; i += 256) s4[i] = g4[i];
    }
#endif
    __syncthreads();

    const int lane = tid & 31;
    const int wave = tid >> 5;     // 0..7 : which 16-edge slab (covers all 128 cols)
    const int lr   = lane & 15;
    const int lh   = lane >> 4;    // 0|1 (upper half-wave)

    const int edgeBase = blockIdx.x * TILE_M;

    // per-lane edge row for A-fragment loads (lanes 16-31 mirror rows 0-15)
    const int arow = edgeBase + wave * 16 + lr;
    const int arc  = arow < E ? arow : E - 1;
    const float* p1 = sites + (size_t)idx1[arc] * S_DIM;
    const float* p2 = sites + (size_t)idx2[arc] * S_DIM;
    const float* pb = bonds + (size_t)arc * BD_DIM;

    v8f acc[16];                   // [ntile*2 + gate]
    #pragma unroll
    for (int i = 0; i < 16; ++i) acc[i] = v8f{0.f,0.f,0.f,0.f,0.f,0.f,0.f,0.f};

    // ---- K loop: 10 steps of 32; segment (s1|s2|bond) is uniform per step ----
    #pragma unroll
    for (int ks = 0; ks < 10; ++ks) {
        const float* base;
        int koff;
        if (ks < 4)      { base = p1; koff = ks * 32; }
        else if (ks < 8) { base = p2; koff = ks * 32 - 128; }
        else             { base = pb; koff = ks * 32 - 256; }

        // A fragment: lanes0-15 K=[k0..k0+15]; upper lanes +16
        const int k0 = koff + lh * 16;
        const float4 f0 = *(const float4*)(base + k0);
        const float4 f1 = *(const float4*)(base + k0 + 4);
        const float4 f2 = *(const float4*)(base + k0 + 8);
        const float4 f3 = *(const float4*)(base + k0 + 12);
        v8u au;
        au[0] = pack2bf(f0.x, f0.y);  au[1] = pack2bf(f0.z, f0.w);
        au[2] = pack2bf(f1.x, f1.y);  au[3] = pack2bf(f1.z, f1.w);
        au[4] = pack2bf(f2.x, f2.y);  au[5] = pack2bf(f2.z, f2.w);
        au[6] = pack2bf(f3.x, f3.y);  au[7] = pack2bf(f3.z, f3.w);
        const v16bf a = __builtin_bit_cast(v16bf, au);

        // B fragments from LDS: lane = 16 contiguous bf16 of its column
        // (2x ds_load_b128); explicit double-buffer so tile nt+1's loads are
        // in flight while tile nt's two WMMAs execute.
        const int kb   = ks * 32 + lh * 16;
        const int cb0  = lr * PADK + kb;              // tile 0 sig offset
        v16bf bs = *(const v16bf*)&smem[cb0];
        v16bf bo = *(const v16bf*)&smem[S_DIM * PADK + cb0];
        #pragma unroll
        for (int nt = 0; nt < 8; ++nt) {
            v16bf ns = bs, no = bo;
            if (nt < 7) {
                const int cbn = ((nt + 1) * 16 + lr) * PADK + kb;
                ns = *(const v16bf*)&smem[cbn];
                no = *(const v16bf*)&smem[S_DIM * PADK + cbn];
            }
            acc[2*nt]   = __builtin_amdgcn_wmma_f32_16x16x32_bf16(
                             false, a, false, bs, (short)0, acc[2*nt],   false, false);
            acc[2*nt+1] = __builtin_amdgcn_wmma_f32_16x16x32_bf16(
                             false, a, false, bo, (short)0, acc[2*nt+1], false, false);
            bs = ns; bo = no;
        }
    }

    // ---- epilogue: bias, sigmoid*relu gate, scatter-add into out[idx1] ----
    int  dsts[8];
    bool live[8];
    #pragma unroll
    for (int v = 0; v < 8; ++v) {
        int row = edgeBase + wave * 16 + lh * 8 + v;   // C layout: M = v + 8*lh
        live[v] = row < E;
        dsts[v] = idx1[row < E ? row : E - 1];
    }
    #pragma unroll
    for (int nt = 0; nt < 8; ++nt) {
        const int n = nt * 16 + lr;                    // C layout: N = lane&15
        const float bv_s = bsig[n];
        const float bv_o = bsoft[n];
        const v8f cs = acc[2*nt];
        const v8f co = acc[2*nt+1];
        #pragma unroll
        for (int v = 0; v < 8; ++v) {
            float x = cs[v] + bv_s;
            float y = co[v] + bv_o;
            float g = (1.0f / (1.0f + __expf(-x))) * (y > 0.0f ? y : 0.0f);
            if (live[v]) atomicAdd(out + (size_t)dsts[v] * S_DIM + n, g);
        }
    }
}

extern "C" void kernel_launch(void* const* d_in, const int* in_sizes, int n_in,
                              void* d_out, int out_size, void* d_ws, size_t ws_size,
                              hipStream_t stream) {
    const float* sites = (const float*)d_in[0];
    const float* bonds = (const float*)d_in[1];
    const float* Wsig  = (const float*)d_in[2];
    const float* bsig  = (const float*)d_in[3];
    const float* Wsoft = (const float*)d_in[4];
    const float* bsoft = (const float*)d_in[5];
    const int*   idx1  = (const int*)d_in[6];
    const int*   idx2  = (const int*)d_in[7];
    float* out = (float*)d_out;
    unsigned* wpack = (unsigned*)d_ws;     // 168 KB packed bf16 weight image

    const int E  = in_sizes[6];            // number of edges
    const int n4 = out_size / 4;           // residual copy in float4s

    init_out_kernel<<<(n4 + 255) / 256, 256, 0, stream>>>(
        (const float4*)sites, (float4*)out, n4);

    prep_weights_kernel<<<(WDWORDS + 255) / 256, 256, 0, stream>>>(Wsig, Wsoft, wpack);

    const int nblocks = (E + TILE_M - 1) / TILE_M;
    cgcnn_edge_wmma<<<nblocks, 256, LDSB, stream>>>(
        sites, bonds, wpack, bsig, bsoft, idx1, idx2, out, E);
}
